// isqrtm_5609227288681
// MI455X (gfx1250) — matrix operand order repaired and checked by reading, not compiled, not run on hardware
//
#include <hip/hip_runtime.h>
#include <hip/hip_bf16.h>
#include <math.h>

typedef __attribute__((ext_vector_type(16))) __bf16 v16bf;
typedef __attribute__((ext_vector_type(8)))  __bf16 v8bf;
typedef __attribute__((ext_vector_type(2)))  __bf16 v2bf;
typedef __attribute__((ext_vector_type(8)))  float  v8f;

#define DIM 64
#define LDA 72   // row stride in bf16 elements; 144B rows -> conflict-free b128 LDS access

__device__ __forceinline__ v8f wmma_bf16(v16bf a, v16bf b, v8f c) {
  // D = A(16x32 bf16) * B(32x16 bf16) + C(16x16 f32)
  return __builtin_amdgcn_wmma_f32_16x16x32_bf16(false, a, false, b, (short)0, c,
                                                 false, false);
}

// A-side fragment: rows m0..m0+15, K = k0..k0+31 of row-major S.
// ISA layout: lane(0-15)=row m, lanes 16-31 same rows; VGPR0-3 hold K=kb..kb+7
// (contiguous -> one b128), VGPR4-7 hold K=kb+16..kb+23 (second b128), kb=(lane>>4)*8.
__device__ __forceinline__ v16bf load_afrag(const __bf16* S, int m0, int k0, int lane) {
  const int m  = m0 + (lane & 15);
  const int kb = k0 + ((lane >> 4) << 3);
  const __bf16* p = S + m * LDA + kb;
  v8bf lo = *(const v8bf*)(p);        // K = kb .. kb+7
  v8bf hi = *(const v8bf*)(p + 16);   // K = kb+16 .. kb+23
  v16bf f;
#pragma unroll
  for (int i = 0; i < 8; ++i) { f[i] = lo[i]; f[i + 8] = hi[i]; }
  return f;
}

// B-side fragment for SYMMETRIC operand: B[k][n] = S[n][k], so column n of B is
// row n of row-major S. lane(0-15)=col n, kb=(lane>>4)*8; VGPR0-3 hold K=kb..kb+7,
// VGPR4-7 hold K=kb+16..kb+23 -> two b128 loads from row n (same order as A).
__device__ __forceinline__ v16bf load_bfrag_sym(const __bf16* S, int n0, int k0, int lane) {
  const int n  = n0 + (lane & 15);
  const int kb = k0 + ((lane >> 4) << 3);
  const __bf16* p = S + n * LDA + kb;
  v8bf lo = *(const v8bf*)(p);
  v8bf hi = *(const v8bf*)(p + 16);
  v16bf f;
#pragma unroll
  for (int i = 0; i < 8; ++i) { f[i] = lo[i]; f[i + 8] = hi[i]; }
  return f;
}

// Wave w computes its 16x64 row stripe of D = P * Q (Q symmetric) into acc[4].
__device__ __forceinline__ void mm_stripe(const __bf16* P, const __bf16* Q,
                                          int w, int lane, v8f acc[4]) {
#pragma unroll
  for (int t = 0; t < 4; ++t)
    acc[t] = (v8f){0.f, 0.f, 0.f, 0.f, 0.f, 0.f, 0.f, 0.f};
#pragma unroll
  for (int ks = 0; ks < 2; ++ks) {
    v16bf a = load_afrag(P, w * 16, ks * 32, lane);   // reused across 4 tiles
#pragma unroll
    for (int t = 0; t < 4; ++t) {
      v16bf b = load_bfrag_sym(Q, t * 16, ks * 32, lane);
      acc[t] = wmma_bf16(a, b, acc[t]);
    }
  }
}

// Write stripe back to LDS as bf16. C/D layout: lane n=(lane&15), VGPR r holds
// row m = w*16 + (lane>>4)*8 + r. We store transposed (pack m,m+1 pairs -> b32
// ds stores); since all iterates are symmetric (polynomials of SPD A), the
// transpose IS the matrix. If zyform: val = 0.5*(3*delta(m,n) - c).
__device__ __forceinline__ void store_stripe(__bf16* S, int w, int lane,
                                             const v8f* acc, bool zyform) {
  const int nl = lane & 15;
  const int m0 = w * 16 + ((lane >> 4) << 3);
#pragma unroll
  for (int t = 0; t < 4; ++t) {
    const int n = t * 16 + nl;
#pragma unroll
    for (int r = 0; r < 8; r += 2) {
      float c0 = acc[t][r], c1 = acc[t][r + 1];
      if (zyform) {
        c0 = ((m0 + r)     == n ? 1.5f : 0.f) - 0.5f * c0;
        c1 = ((m0 + r + 1) == n ? 1.5f : 0.f) - 0.5f * c1;
      }
      v2bf p2; p2[0] = (__bf16)c0; p2[1] = (__bf16)c1;
      *(v2bf*)(S + n * LDA + m0 + r) = p2;   // 4B-aligned packed store
    }
  }
}

__device__ __forceinline__ void store_out(float* __restrict__ ob, int w, int lane,
                                          const v8f* acc, float s) {
  const int nl = lane & 15;
  const int m0 = w * 16 + ((lane >> 4) << 3);
#pragma unroll
  for (int t = 0; t < 4; ++t) {
    const int n = t * 16 + nl;
#pragma unroll
    for (int r = 0; r < 8; ++r)
      ob[(m0 + r) * DIM + n] = acc[t][r] * s;
  }
}

__global__ __launch_bounds__(128) void isqrtm_ns_kernel(const float* __restrict__ x,
                                                        const int* __restrict__ iterN_p,
                                                        float* __restrict__ out) {
  __shared__ __attribute__((aligned(16))) __bf16 sY [DIM * LDA];
  __shared__ __attribute__((aligned(16))) __bf16 sZ [DIM * LDA];
  __shared__ __attribute__((aligned(16))) __bf16 sZY[DIM * LDA];
  __shared__ float sdiag[DIM];
  __shared__ float snorm;

  const int tid  = threadIdx.x;
  const int lane = tid & 31;
  const int w    = tid >> 5;
  const float* __restrict__ xb = x   + (size_t)blockIdx.x * (DIM * DIM);
  float* __restrict__       ob = out + (size_t)blockIdx.x * (DIM * DIM);

  // ---- normA = trace(x) ----
  if (tid < DIM) sdiag[tid] = xb[tid * (DIM + 1)];
  __syncthreads();
  if (tid == 0) {
    float s = 0.f;
    for (int i = 0; i < DIM; ++i) s += sdiag[i];
    snorm = s;
  }
  __syncthreads();
  const float norm = snorm;
  const float inv  = 1.0f / norm;
  const float sn   = sqrtf(norm);

  // ---- A = x/normA -> sY ; ZY0 = 0.5*(3I - A) -> sZ (Z starts as ZY0) ----
  for (int i = tid; i < (DIM * DIM) / 4; i += 128) {
    float4 v = ((const float4*)xb)[i];
    const int e = i * 4, m = e >> 6, n = e & 63;
    const float a[4] = {v.x * inv, v.y * inv, v.z * inv, v.w * inv};
    __bf16* py = sY + m * LDA + n;
    __bf16* pz = sZ + m * LDA + n;
#pragma unroll
    for (int j = 0; j < 4; ++j) {
      py[j] = (__bf16)a[j];
      pz[j] = (__bf16)(0.5f * (((n + j) == m ? 3.f : 0.f) - a[j]));
    }
  }
  __syncthreads();

  const int iterN = iterN_p[0];

  // ---- Y = A @ ZY0 ----
  v8f acc[4];
  mm_stripe(sY, sZ, w, lane, acc);
  if (iterN < 2) {               // reference's early-out == A @ 0.5*(3I - A)
    store_out(ob, w, lane, acc, sn);
    return;
  }
  __syncthreads();               // all reads of sY (=A) done
  store_stripe(sY, w, lane, acc, false);   // sY = Y0, sZ = Z0 = ZY0
  __syncthreads();

  // ---- Newton-Schulz: (iterN-1) rounds; last round skips the Z update ----
  const int nloop = iterN - 1;
  for (int it = 0; it < nloop; ++it) {
    const bool last = (it == nloop - 1);

    mm_stripe(sZ, sY, w, lane, acc);          // T = Z @ Y
    __syncthreads();
    store_stripe(sZY, w, lane, acc, true);    // ZY = 0.5*(3I - T)
    __syncthreads();

    v8f accY[4];
    mm_stripe(sY, sZY, w, lane, accY);        // Ynew = Y @ ZY
    if (last) {
      store_out(ob, w, lane, accY, sn);       // y = Y * sqrt(normA)
      return;
    }
    v8f accZ[4];
    mm_stripe(sZY, sZ, w, lane, accZ);        // Znew = ZY @ Z
    __syncthreads();                          // reads of old sY/sZ done
    store_stripe(sY, w, lane, accY, false);
    store_stripe(sZ, w, lane, accZ, false);
    __syncthreads();
  }
}

extern "C" void kernel_launch(void* const* d_in, const int* in_sizes, int n_in,
                              void* d_out, int out_size, void* d_ws, size_t ws_size,
                              hipStream_t stream) {
  (void)n_in; (void)out_size; (void)d_ws; (void)ws_size;
  const float* x  = (const float*)d_in[0];
  const int* itp  = (const int*)d_in[1];
  float* out      = (float*)d_out;
  const int batches = in_sizes[0] / (DIM * DIM);
  isqrtm_ns_kernel<<<dim3(batches), dim3(128), 0, stream>>>(x, itp, out);
}
